// AttentionHead_64476049047604
// MI455X (gfx1250) — matrix operand order repaired and checked by reading, not compiled, not run on hardware
//
#include <hip/hip_runtime.h>
#include <hip/hip_bf16.h>

typedef __attribute__((ext_vector_type(16))) _Float16 v16h;
typedef __attribute__((ext_vector_type(8)))  _Float16 v8h;
typedef __attribute__((ext_vector_type(8)))  float    v8f;

#define B_   2
#define S_   2048
#define D_   1024
#define H_   16
#define HD_  64
#define ROWQKV 3072   // Q(1024) | K(1024) | V(1024) per token row

// ---------------- WMMA helpers (CDNA5 16x16x32 f16, f32 accum) ----------------
__device__ __forceinline__ v8f wmma16(v16h a, v16h b, v8f c) {
  // 8 args: (neg_a, A, neg_b, B, c_mod, C, reuse_a, reuse_b)
  return __builtin_amdgcn_wmma_f32_16x16x32_f16(false, a, false, b, (short)0, c,
                                                false, false);
}

// A-fragment (16x32, 16-bit): lane L -> M = L%16; K = {kb..kb+15},
// kb = 16*(L/16). Two contiguous 16B chunks from the row.
__device__ __forceinline__ v16h load_a_frag(const _Float16* row, int kb) {
  v8h lo = *(const v8h*)(row + kb);
  v8h hi = *(const v8h*)(row + kb + 8);
  return __builtin_shufflevector(lo, hi, 0, 1, 2, 3, 4, 5, 6, 7,
                                 8, 9, 10, 11, 12, 13, 14, 15);
}

// Half-wave (16-lane) reductions: wave32, rows split lanes 0-15 / 16-31.
__device__ __forceinline__ float half_max16(float v) {
  #pragma unroll
  for (int off = 1; off < 16; off <<= 1) v = fmaxf(v, __shfl_xor(v, off, 16));
  return v;
}
__device__ __forceinline__ float half_sum16(float v) {
  #pragma unroll
  for (int off = 1; off < 16; off <<= 1) v += __shfl_xor(v, off, 16);
  return v;
}

// ---------------- small prep kernels ----------------
__global__ void f32_to_f16_kernel(const float* __restrict__ src,
                                  _Float16* __restrict__ dst, int n) {
  int i = blockIdx.x * blockDim.x + threadIdx.x;
  if (i < n) dst[i] = (_Float16)src[i];
}

// src: K x N (row-major, f32)  ->  dst: N x K (row-major, f16)   (BT layout)
__global__ void transpose_w_kernel(const float* __restrict__ src,
                                   _Float16* __restrict__ dst, int K, int N) {
  int i = blockIdx.x * blockDim.x + threadIdx.x;
  if (i < K * N) {
    int n = i / K;
    int k = i - n * K;
    dst[i] = (_Float16)src[(size_t)k * N + n];
  }
}

__global__ void pack_bias_kernel(const float* __restrict__ bq,
                                 const float* __restrict__ bkv,
                                 float* __restrict__ dst) {
  int i = blockIdx.x * blockDim.x + threadIdx.x;
  if (i < 1024) dst[i] = bq[i];
  else if (i < 3072) dst[i] = bkv[i - 1024];
}

// V part of qkv16 -> per-head transposed Vt[(b*H+h)*HD + d][m]  (m contiguous)
__global__ void transpose_v_kernel(const _Float16* __restrict__ qkv,
                                   _Float16* __restrict__ vT) {
  int i = blockIdx.x * blockDim.x + threadIdx.x;   // over B*H*HD*S = 4194304
  if (i >= B_ * H_ * HD_ * S_) return;
  int m  = i & (S_ - 1);
  int d  = (i >> 11) & (HD_ - 1);
  int bh = i >> 17;
  int b = bh >> 4, h = bh & 15;
  vT[i] = qkv[(size_t)(b * S_ + m) * ROWQKV + 2048 + h * HD_ + d];
}

// ---------------- tiled WMMA GEMM: C = A(MxK,f16) * BT(NxK,f16)^T + bias ------
template <bool F16OUT>
__global__ __launch_bounds__(128) void gemm_wmma_kernel(
    const _Float16* __restrict__ A, const _Float16* __restrict__ BT,
    const float* __restrict__ bias, void* __restrict__ Cout,
    int M, int N, int K, int ldc) {
  int lane = threadIdx.x & 31;
  int w = (blockIdx.x * blockDim.x + threadIdx.x) >> 5;
  int nt = N >> 5;
  int wm = w / nt, wn = w - wm * nt;
  if (wm >= (M >> 5)) return;
  int m0 = wm << 5, n0 = wn << 5;
  int r = lane & 15, hi = lane >> 4;
  int kb = hi ? 16 : 0;       // A-fragment chunk base
  int kb2 = hi << 4;          // B-fragment K base (0 or 16)

  const _Float16* arow0 = A + (size_t)(m0 + r) * K;
  const _Float16* arow1 = A + (size_t)(m0 + 16 + r) * K;
  const _Float16* brow0 = BT + (size_t)(n0 + r) * K + kb2;
  const _Float16* brow1 = BT + (size_t)(n0 + 16 + r) * K + kb2;

  v8f acc00 = {}, acc01 = {}, acc10 = {}, acc11 = {};
  for (int k0 = 0; k0 < K; k0 += 32) {
    v16h a0 = load_a_frag(arow0 + k0, kb);
    v16h a1 = load_a_frag(arow1 + k0, kb);
    v16h b0 = *(const v16h*)(brow0 + k0);
    v16h b1 = *(const v16h*)(brow1 + k0);
    acc00 = wmma16(a0, b0, acc00);
    acc01 = wmma16(a0, b1, acc01);
    acc10 = wmma16(a1, b0, acc10);
    acc11 = wmma16(a1, b1, acc11);
  }

  float bv0 = bias[n0 + r], bv1 = bias[n0 + 16 + r];
  #pragma unroll
  for (int v = 0; v < 8; ++v) {
    int row0 = m0 + v + 8 * hi, row1 = row0 + 16;
    int col0 = n0 + r, col1 = col0 + 16;
    float e00 = acc00[v] + bv0, e01 = acc01[v] + bv1;
    float e10 = acc10[v] + bv0, e11 = acc11[v] + bv1;
    if (F16OUT) {
      _Float16* C = (_Float16*)Cout;
      C[(size_t)row0 * ldc + col0] = (_Float16)e00;
      C[(size_t)row0 * ldc + col1] = (_Float16)e01;
      C[(size_t)row1 * ldc + col0] = (_Float16)e10;
      C[(size_t)row1 * ldc + col1] = (_Float16)e11;
    } else {
      float* C = (float*)Cout;
      C[(size_t)row0 * ldc + col0] = e00;
      C[(size_t)row0 * ldc + col1] = e01;
      C[(size_t)row1 * ldc + col0] = e10;
      C[(size_t)row1 * ldc + col1] = e11;
    }
  }
}

// ---------------- attention: one wave per (b, h, 16 query rows) ---------------
// 3 WMMA passes over K-rows: (1) row max, (2) row sum, (3) normalize+write
// weights (the single 512MB write) and accumulate P*V via LDS re-layout.
__global__ __launch_bounds__(128) void attn_kernel(
    const _Float16* __restrict__ qkv, const _Float16* __restrict__ vT,
    float* __restrict__ wout, _Float16* __restrict__ attn16) {
  __shared__ __align__(32) _Float16 pb[4][16 * 32];   // 1KB P tile per wave
  const float scale = 0.125f;                         // hd^-0.5
  int lane = threadIdx.x & 31;
  int wib = threadIdx.x >> 5;
  int w = blockIdx.x * 4 + wib;                       // 4096 waves total
  int b = w >> 11;
  int h = (w >> 7) & 15;
  int n0 = (w & 127) << 4;
  int r = lane & 15, hi = lane >> 4;
  int kb = hi ? 16 : 0;
  int kb2 = hi << 4;

  // Q A-fragments for rows n0..n0+15, k=0..63 (held across all passes)
  const _Float16* qrow = qkv + (size_t)(b * S_ + n0 + r) * ROWQKV + h * HD_;
  v16h qa0 = load_a_frag(qrow, kb);
  v16h qa1 = load_a_frag(qrow + 32, kb);
  const _Float16* kbase = qkv + (size_t)b * S_ * ROWQKV + 1024 + h * HD_ + kb2;

  // ---- pass 1: row max (track raw, scale once; scale > 0) ----
  float rm[8];
  #pragma unroll
  for (int v = 0; v < 8; ++v) rm[v] = -3.0e38f;
  for (int m0 = 0; m0 < S_; m0 += 16) {
    const _Float16* kr = kbase + (size_t)(m0 + r) * ROWQKV;
    v16h b0 = *(const v16h*)(kr);
    v16h b1 = *(const v16h*)(kr + 32);
    v8f c = {};
    c = wmma16(qa0, b0, c);
    c = wmma16(qa1, b1, c);
    #pragma unroll
    for (int v = 0; v < 8; ++v) rm[v] = fmaxf(rm[v], c[v]);
  }
  #pragma unroll
  for (int v = 0; v < 8; ++v) rm[v] = half_max16(rm[v]) * scale;

  // ---- pass 2: row sum of exp ----
  float rs[8] = {0.f, 0.f, 0.f, 0.f, 0.f, 0.f, 0.f, 0.f};
  for (int m0 = 0; m0 < S_; m0 += 16) {
    const _Float16* kr = kbase + (size_t)(m0 + r) * ROWQKV;
    v16h b0 = *(const v16h*)(kr);
    v16h b1 = *(const v16h*)(kr + 32);
    v8f c = {};
    c = wmma16(qa0, b0, c);
    c = wmma16(qa1, b1, c);
    #pragma unroll
    for (int v = 0; v < 8; ++v) rs[v] += __expf(c[v] * scale - rm[v]);
  }
  float inv[8];
  #pragma unroll
  for (int v = 0; v < 8; ++v) inv[v] = 1.0f / half_sum16(rs[v]);

  // ---- pass 3: write normalized weights + accumulate O = P*V ----
  v8f o0 = {}, o1 = {}, o2 = {}, o3 = {};
  _Float16* myp = pb[wib];
  const _Float16* vbase = vT + (size_t)((b * H_ + h) * HD_ + r) * S_ + kb2;
  for (int mb = 0; mb < S_; mb += 32) {
    #pragma unroll
    for (int t = 0; t < 2; ++t) {
      int m0 = mb + t * 16;
      const _Float16* kr = kbase + (size_t)(m0 + r) * ROWQKV;
      v16h b0 = *(const v16h*)(kr);
      v16h b1 = *(const v16h*)(kr + 32);
      v8f c = {};
      c = wmma16(qa0, b0, c);
      c = wmma16(qa1, b1, c);
      #pragma unroll
      for (int v = 0; v < 8; ++v) {
        float p = __expf(c[v] * scale - rm[v]);
        int M = v + 8 * hi;
        myp[M * 32 + t * 16 + r] = (_Float16)p;   // stage unnormalized P
        // weights[b, n, m, h]  (single pass over the 512MB tensor)
        wout[(((size_t)(b * S_ + n0 + M)) * S_ + (m0 + r)) * H_ + h] = p * inv[v];
      }
    }
    __syncthreads();   // C-layout -> A-fragment re-layout through LDS
    v16h pa = load_a_frag(myp + r * 32, kb);
    o0 = wmma16(pa, *(const v16h*)(vbase + (size_t)0 * 16 * S_ + mb), o0);
    o1 = wmma16(pa, *(const v16h*)(vbase + (size_t)1 * 16 * S_ + mb), o1);
    o2 = wmma16(pa, *(const v16h*)(vbase + (size_t)2 * 16 * S_ + mb), o2);
    o3 = wmma16(pa, *(const v16h*)(vbase + (size_t)3 * 16 * S_ + mb), o3);
    __syncthreads();
  }

  #pragma unroll
  for (int v = 0; v < 8; ++v) {
    int M = v + 8 * hi;
    _Float16* orow = attn16 + (size_t)(b * S_ + n0 + M) * D_ + h * HD_ + r;
    orow[0]  = (_Float16)(o0[v] * inv[v]);
    orow[16] = (_Float16)(o1[v] * inv[v]);
    orow[32] = (_Float16)(o2[v] * inv[v]);
    orow[48] = (_Float16)(o3[v] * inv[v]);
  }
}

// ---------------- host side ----------------
extern "C" void kernel_launch(void* const* d_in, const int* in_sizes, int n_in,
                              void* d_out, int out_size, void* d_ws, size_t ws_size,
                              hipStream_t stream) {
  (void)in_sizes; (void)n_in; (void)out_size; (void)ws_size;
  const float* queries = (const float*)d_in[0];
  const float* Wq  = (const float*)d_in[1];
  const float* bq  = (const float*)d_in[2];
  const float* Wkv = (const float*)d_in[3];
  const float* bkv = (const float*)d_in[4];
  const float* Wo  = (const float*)d_in[5];
  const float* bo  = (const float*)d_in[6];

  char* ws = (char*)d_ws;
  const size_t MiB = 1u << 20;
  _Float16* x16   = (_Float16*)(ws + 0 * MiB);    // 8 MiB  : X f16
  _Float16* wT    = (_Float16*)(ws + 8 * MiB);    // 6 MiB  : [Wq|Wkv]^T f16
  _Float16* woT   = (_Float16*)(ws + 14 * MiB);   // 2 MiB  : Wo^T f16
  float*    biasA = (float*)   (ws + 16 * MiB);   // 12 KiB : [bq|bkv]
  _Float16* qkv16 = (_Float16*)(ws + 17 * MiB);   // 24 MiB : Q|K|V f16
  _Float16* vT16  = (_Float16*)(ws + 41 * MiB);   // 8 MiB  : V^T per head
  _Float16* attn16= (_Float16*)(ws + 49 * MiB);   // 8 MiB  : attention out f16

  // 1) precision convert + weight transposes (BT layout for WMMA B-fragments)
  f32_to_f16_kernel<<<16384, 256, 0, stream>>>(queries, x16, B_ * S_ * D_);
  transpose_w_kernel<<<4096, 256, 0, stream>>>(Wq, wT, 1024, 1024);
  transpose_w_kernel<<<8192, 256, 0, stream>>>(Wkv, wT + 1024 * 1024, 1024, 2048);
  transpose_w_kernel<<<4096, 256, 0, stream>>>(Wo, woT, 1024, 1024);
  pack_bias_kernel<<<12, 256, 0, stream>>>(bq, bkv, biasA);

  // 2) fused QKV projection: (4096x1024) x (1024x3072) -> f16
  gemm_wmma_kernel<true><<<3072, 128, 0, stream>>>(
      x16, wT, biasA, (void*)qkv16, 4096, 3072, 1024, ROWQKV);

  // 3) per-head V transpose (m-contiguous rows for PV B-fragments)
  transpose_v_kernel<<<16384, 256, 0, stream>>>(qkv16, vT16);

  // 4) attention: softmax weights (written once) + P*V
  float* out = (float*)d_out;
  float* weights = out + (size_t)B_ * S_ * D_;    // tuple: (out, weights)
  attn_kernel<<<1024, 128, 0, stream>>>(qkv16, vT16, weights, attn16);

  // 5) output projection: (4096x1024) x (1024x1024) + bo -> f32 out
  gemm_wmma_kernel<false><<<1024, 128, 0, stream>>>(
      attn16, woT, bo, d_out, 4096, 1024, 1024, 1024);
}
